// cWCT_8950711845891
// MI455X (gfx1250) — matrix operand order repaired and checked by reading, not compiled, not run on hardware
//
#include <hip/hip_runtime.h>

#define BATCH 8
#define CH    128
#define CH2   256
#define IMH   256
#define IMW   256
#define NPIX  65536
#define OHW   128
#define OPIX  16384
#define KSPLIT 32
#define KRANGE 2048          // NPIX / KSPLIT
#define EPSWCT 2e-5f
#define GSTRIDE 40           // bf16 LDS row stride (80B, 16B-aligned)
#define ASTRIDE 136          // bf16 LDS row stride (272B, 16B-aligned)
#define P1W 132              // conv1 panel px dim (130 used)
#define P2W 36               // conv2 panel px dim (34 used)

typedef __attribute__((ext_vector_type(16))) __bf16 v16bf;
typedef __attribute__((ext_vector_type(8)))  __bf16 v8bf;
typedef __attribute__((ext_vector_type(8)))  float  v8f;

__device__ __forceinline__ unsigned short f2bf(float f) {
  union { float f; unsigned u; } v; v.f = f;
  unsigned r = v.u + 0x7FFFu + ((v.u >> 16) & 1u);   // round-to-nearest-even
  return (unsigned short)(r >> 16);
}

__device__ __forceinline__ v16bf mk16(const unsigned short* lo, const unsigned short* hi) {
  v8bf l = *(const v8bf*)lo;
  v8bf h = *(const v8bf*)hi;
  return __builtin_shufflevector(l, h, 0,1,2,3,4,5,6,7,8,9,10,11,12,13,14,15);
}

// pack 16 f32 -> 16 bf16, store as two b128 LDS writes
__device__ __forceinline__ void st_bf16x16(unsigned short* lr,
                                           float4 f0, float4 f1, float4 f2, float4 f3) {
  union { unsigned short s[8]; uint4 v; } a, b;
  a.s[0]=f2bf(f0.x); a.s[1]=f2bf(f0.y); a.s[2]=f2bf(f0.z); a.s[3]=f2bf(f0.w);
  a.s[4]=f2bf(f1.x); a.s[5]=f2bf(f1.y); a.s[6]=f2bf(f1.z); a.s[7]=f2bf(f1.w);
  b.s[0]=f2bf(f2.x); b.s[1]=f2bf(f2.y); b.s[2]=f2bf(f2.z); b.s[3]=f2bf(f2.w);
  b.s[4]=f2bf(f3.x); b.s[5]=f2bf(f3.y); b.s[6]=f2bf(f3.z); b.s[7]=f2bf(f3.w);
  *(uint4*)lr = a.v;
  *(uint4*)(lr + 8) = b.v;
}

#define WMMA_BF16(Aop, Bop, Cacc) \
  __builtin_amdgcn_wmma_f32_16x16x32_bf16(false, (Aop), false, (Bop), (short)0, (Cacc), false, false)

// ---------------------------------------------------------------------------
// Kernel 1: repack conv weights OIHW f32 -> [tap][O][I] bf16
// ---------------------------------------------------------------------------
__global__ __launch_bounds__(256) void repack_kernel(
    const float* __restrict__ W1, const float* __restrict__ W2,
    unsigned short* __restrict__ w1bf, unsigned short* __restrict__ w2bf) {
  int idx = blockIdx.x * 256 + threadIdx.x;
  if (idx < 9 * CH * CH) {
    int t = idx / (CH * CH), r = idx % (CH * CH);
    int o = r >> 7, i = r & 127;
    w1bf[idx] = f2bf(W1[(o * CH + i) * 9 + t]);
  }
  if (idx < 9 * CH2 * CH) {
    int t = idx / (CH2 * CH), r = idx % (CH2 * CH);
    int o = r >> 7, i = r & 127;
    w2bf[idx] = f2bf(W2[(o * CH + i) * 9 + t]);
  }
}

// ---------------------------------------------------------------------------
// Kernel 2: per-(batch,channel) means of content and style features
// ---------------------------------------------------------------------------
__global__ __launch_bounds__(256) void mean_kernel(
    const float* __restrict__ xc, const float* __restrict__ xs,
    float* __restrict__ mc, float* __restrict__ ms) {
  int id = blockIdx.x;                               // [0, 2*B*C)
  const float* src; float* dst; int bc;
  if (id < BATCH * CH) { src = xc; dst = mc; bc = id; }
  else                 { src = xs; dst = ms; bc = id - BATCH * CH; }
  const float4* p = (const float4*)(src + (size_t)bc * NPIX);
  float s = 0.f;
  for (int i = threadIdx.x; i < NPIX / 4; i += 256) {
    float4 f = p[i];
    s += f.x + f.y + f.z + f.w;
  }
  __shared__ float red[256];
  red[threadIdx.x] = s;
  __syncthreads();
  for (int o = 128; o > 0; o >>= 1) {
    if (threadIdx.x < o) red[threadIdx.x] += red[threadIdx.x + o];
    __syncthreads();
  }
  if (threadIdx.x == 0) dst[bc] = red[0] * (1.0f / NPIX);
}

// ---------------------------------------------------------------------------
// Kernel 3: split-K Gram  G = X * X^T  (128x128, K=65536) via bf16 WMMA.
// Double-buffered LDS staging (1 barrier / K-step); operand loads hoisted so
// the 8 WMMAs issue back-to-back behind a single s_wait_dscnt.
// ---------------------------------------------------------------------------
__global__ __launch_bounds__(256) void gram_kernel(
    const float* __restrict__ xc, const float* __restrict__ xs,
    float* __restrict__ gpart) {
  __shared__ alignas(16) unsigned short lds[2][CH * GSTRIDE];
  int ks = blockIdx.x;
  int mi = blockIdx.y;
  const float* X = (mi < 8) ? xc + (size_t)mi * CH * NPIX
                            : xs + (size_t)(mi - 8) * CH * NPIX;
  int k0 = ks * KRANGE;
  int tid = threadIdx.x, wave = tid >> 5, lane = tid & 31;
  int ln = lane & 15, kh = lane >> 4;
  v8f acc[8] = {};
  int row = tid >> 1, seg = (tid & 1) * 16;
  const float* base = X + (size_t)row * NPIX + seg;
  unsigned short* myslot0 = &lds[0][row * GSTRIDE + seg];
  unsigned short* myslot1 = &lds[1][row * GSTRIDE + seg];
  {
    const float4* gp = (const float4*)(base + k0);
    st_bf16x16(myslot0, gp[0], gp[1], gp[2], gp[3]);
  }
  __syncthreads();
  const int NIT = KRANGE / 32;                       // 64
  for (int it = 0; it < NIT; ++it) {
    int cur = it & 1;
    bool has = (it + 1) < NIT;
    float4 g0, g1, g2, g3;
    if (has) {                                       // issue next-chunk loads early
      const float4* gp = (const float4*)(base + k0 + (it + 1) * 32);
      g0 = gp[0]; g1 = gp[1]; g2 = gp[2]; g3 = gp[3];
      __builtin_prefetch((const void*)(gp + 8), 0, 3);
    }
    const unsigned short* ar = &lds[cur][(wave * 16 + ln) * GSTRIDE];
    v16bf A = mk16(ar + kh * 16, ar + kh * 16 + 8);
    v16bf Bv[8];
    #pragma unroll
    for (int nt = 0; nt < 8; ++nt) {
      const unsigned short* br = &lds[cur][(nt * 16 + ln) * GSTRIDE + kh * 16];
      Bv[nt] = mk16(br, br + 8);
    }
    #pragma unroll
    for (int nt = 0; nt < 8; ++nt) acc[nt] = WMMA_BF16(A, Bv[nt], acc[nt]);
    if (has) st_bf16x16(cur ? myslot0 : myslot1, g0, g1, g2, g3);
    __syncthreads();
  }
  float* G = gpart + ((size_t)mi * KSPLIT + ks) * (CH * CH);
  int mb = kh * 8;
  #pragma unroll
  for (int nt = 0; nt < 8; ++nt)
    #pragma unroll
    for (int v = 0; v < 8; ++v)
      G[(size_t)(wave * 16 + mb + v) * CH + nt * 16 + ln] = acc[nt][v];
}

// ---------------------------------------------------------------------------
// Kernel 4: deterministic reduction of split-K Gram partials
// ---------------------------------------------------------------------------
__global__ __launch_bounds__(256) void gram_reduce_kernel(
    const float* __restrict__ gpart, float* __restrict__ gc, float* __restrict__ gs) {
  int idx = blockIdx.x * 256 + threadIdx.x;          // 16 * 16384
  int mi = idx >> 14, e = idx & 16383;
  const float* p = gpart + ((size_t)mi * KSPLIT) * (CH * CH) + e;
  float s = 0.f;
  #pragma unroll 4
  for (int k = 0; k < KSPLIT; ++k) s += p[(size_t)k * CH * CH];
  if (mi < 8) gc[(size_t)mi * CH * CH + e] = s;
  else        gs[(size_t)(mi - 8) * CH * CH + e] = s;
}

// ---------------------------------------------------------------------------
// Kernel 5: per batch -- cov correction, Cholesky, tri-inverse, A = Ls*invLc,
// d = ms - A*mc.  One block per batch, 130KB dynamic LDS.
// ---------------------------------------------------------------------------
__device__ void load_cov(float* Lb, const float* __restrict__ G, const float* mean, int tid) {
  const float scale = 1.0f / (float)(NPIX - 1);
  for (int idx = tid; idx < CH * CH; idx += 256) {
    int i = idx >> 7, j = idx & 127;
    float v = (G[idx] - (float)NPIX * mean[i] * mean[j]) * scale;
    if (i == j) v += EPSWCT;
    Lb[i * 129 + j] = v;
  }
  __syncthreads();
}

__device__ void cholesky128(float* Lb, int tid) {
  for (int j = 0; j < CH; ++j) {
    if (tid == 0) Lb[j * 129 + j] = sqrtf(Lb[j * 129 + j]);
    __syncthreads();
    float inv_d = 1.0f / Lb[j * 129 + j];
    for (int i = j + 1 + tid; i < CH; i += 256) Lb[i * 129 + j] *= inv_d;
    __syncthreads();
    for (int i = j + 1 + tid; i < CH; i += 256) {
      float lij = Lb[i * 129 + j];
      for (int l = j + 1; l <= i; ++l) Lb[i * 129 + l] -= lij * Lb[l * 129 + j];
    }
    __syncthreads();
  }
}

__device__ void trinv128(const float* Lb, float* Inv, int tid) {
  if (tid < CH) {
    int c = tid;
    for (int i = 0; i < c; ++i) Inv[i * 129 + c] = 0.0f;
    Inv[c * 129 + c] = 1.0f / Lb[c * 129 + c];
    for (int i = c + 1; i < CH; ++i) {
      float s = 0.f;
      for (int k = c; k < i; ++k) s += Lb[i * 129 + k] * Inv[k * 129 + c];
      Inv[i * 129 + c] = -s / Lb[i * 129 + i];
    }
  }
  __syncthreads();
}

__global__ __launch_bounds__(256) void chol_kernel(
    const float* __restrict__ gc, const float* __restrict__ gs,
    const float* __restrict__ mc, const float* __restrict__ ms,
    float* __restrict__ Amat, float* __restrict__ dvec) {
  extern __shared__ float4 smem_f4[];
  float* L1  = (float*)smem_f4;          // [128][129] cov / L
  float* L2  = L1 + CH * 129;            // [128][129] inv(Lc)
  float* tv  = L2 + CH * 129;            // [128]
  float* mcs = tv + CH;                  // [128]
  float* mss = mcs + CH;                 // [128]
  int b = blockIdx.x, tid = threadIdx.x;
  if (tid < CH) { mcs[tid] = mc[b * CH + tid]; mss[tid] = ms[b * CH + tid]; }
  __syncthreads();
  load_cov(L1, gc + (size_t)b * CH * CH, mcs, tid);
  cholesky128(L1, tid);
  trinv128(L1, L2, tid);
  load_cov(L1, gs + (size_t)b * CH * CH, mss, tid);   // L1 := Ls
  cholesky128(L1, tid);
  float* Ab = Amat + (size_t)b * CH * CH;
  for (int idx = tid; idx < CH * CH; idx += 256) {    // A = Ls * invLc (lower tri)
    int i = idx >> 7, j = idx & 127;
    float s = 0.f;
    if (i >= j) for (int k = j; k <= i; ++k) s += L1[i * 129 + k] * L2[k * 129 + j];
    Ab[idx] = s;
  }
  if (tid < CH) {                                     // tv = invLc * mc
    float s = 0.f;
    for (int j = 0; j <= tid; ++j) s += L2[tid * 129 + j] * mcs[j];
    tv[tid] = s;
  }
  __syncthreads();
  if (tid < CH) {                                     // d = ms - Ls*tv
    float s = 0.f;
    for (int k = 0; k <= tid; ++k) s += L1[tid * 129 + k] * tv[k];
    dvec[b * CH + tid] = mss[tid] - s;
  }
}

// ---------------------------------------------------------------------------
// Kernel 6: colored = A @ xc + d   ->  bf16.  Tile: 128 ch x 128 px / block.
// ---------------------------------------------------------------------------
__global__ __launch_bounds__(256) void color_kernel(
    const float* __restrict__ xc, const float* __restrict__ Amat,
    const float* __restrict__ dvec, unsigned short* __restrict__ colored) {
  extern __shared__ float4 smem_f4[];
  unsigned short* Al   = (unsigned short*)smem_f4;   // [CH][ASTRIDE]
  unsigned short* Xt   = Al + CH * ASTRIDE;          // [128 px][ASTRIDE]  (transposed x)
  unsigned short* OutS = Xt + CH * ASTRIDE;          // [CH][128]
  int p0 = blockIdx.x * 128, b = blockIdx.y;
  int tid = threadIdx.x, wave = tid >> 5, lane = tid & 31;
  int ln = lane & 15, kh = lane >> 4;
  const float* Ab = Amat + (size_t)b * CH * CH;
  const float* Xb = xc + (size_t)b * CH * NPIX;
  const float* dv = dvec + b * CH;
  for (int i = tid; i < CH * CH / 2; i += 256) {     // A -> bf16 LDS (paired)
    int r = i >> 6, c2 = (i & 63) * 2;
    float2 f = *(const float2*)(Ab + r * CH + c2);
    unsigned pk = (unsigned)f2bf(f.x) | ((unsigned)f2bf(f.y) << 16);
    *(unsigned*)(Al + r * ASTRIDE + c2) = pk;
  }
  {
    int c = tid >> 1, ps = (tid & 1) * 64;
    const float4* xp = (const float4*)(Xb + (size_t)c * NPIX + p0 + ps);
    #pragma unroll
    for (int q = 0; q < 16; ++q) {
      float4 f = xp[q];
      int p = ps + 4 * q;
      Xt[(p + 0) * ASTRIDE + c] = f2bf(f.x);
      Xt[(p + 1) * ASTRIDE + c] = f2bf(f.y);
      Xt[(p + 2) * ASTRIDE + c] = f2bf(f.z);
      Xt[(p + 3) * ASTRIDE + c] = f2bf(f.w);
    }
  }
  __syncthreads();
  v8f acc[8] = {};
  const unsigned short* ar = Al + (wave * 16 + ln) * ASTRIDE;
  #pragma unroll
  for (int ksi = 0; ksi < 4; ++ksi) {
    int k = ksi * 32;
    v16bf A = mk16(ar + k + kh * 16, ar + k + kh * 16 + 8);
    v16bf Bv[8];
    #pragma unroll
    for (int nt = 0; nt < 8; ++nt) {
      const unsigned short* br = Xt + (nt * 16 + ln) * ASTRIDE + k + kh * 16;
      Bv[nt] = mk16(br, br + 8);
    }
    #pragma unroll
    for (int nt = 0; nt < 8; ++nt) acc[nt] = WMMA_BF16(A, Bv[nt], acc[nt]);
  }
  int mb = kh * 8;
  #pragma unroll
  for (int nt = 0; nt < 8; ++nt)
    #pragma unroll
    for (int v = 0; v < 8; ++v) {
      int c = wave * 16 + mb + v;
      OutS[c * 128 + nt * 16 + ln] = f2bf(acc[nt][v] + dv[c]);
    }
  __syncthreads();
  {
    int c = tid >> 1, seg = (tid & 1) * 64;
    const uint4* s4 = (const uint4*)(OutS + c * 128 + seg);
    uint4* d4 = (uint4*)(colored + ((size_t)b * CH + c) * NPIX + p0 + seg);
    #pragma unroll
    for (int q = 0; q < 8; ++q) d4[q] = s4[q];
  }
}

// ---------------------------------------------------------------------------
// Kernel 7: reflect-pad + conv 3x3 stride2 + bias + ReLU -> bf16.
// 9 tap-GEMMs; tile 128 out-ch x 64 out-px per block.
// ---------------------------------------------------------------------------
__global__ __launch_bounds__(256) void conv1_kernel(
    const unsigned short* __restrict__ colored,
    const unsigned short* __restrict__ w1bf,
    const float* __restrict__ b1, unsigned short* __restrict__ x1) {
  extern __shared__ float4 smem_f4[];
  unsigned short* P    = (unsigned short*)smem_f4;   // [3][P1W][ASTRIDE] px-major panels
  unsigned short* Atap = P + 3 * P1W * ASTRIDE;      // [CH][ASTRIDE]
  unsigned short* S    = Atap + CH * ASTRIDE;        // [CH][64]
  int x0 = blockIdx.x * 64, oy = blockIdx.y, b = blockIdx.z;
  int tid = threadIdx.x, wave = tid >> 5, lane = tid & 31;
  int ln = lane & 15, kh = lane >> 4;
  const unsigned short* src = colored + (size_t)b * CH * NPIX;
  {
    int c = tid & 127, g = tid >> 7;
    for (int pr = 0; pr < 3; ++pr) {
      int iy = 2 * oy + pr - 1;
      iy = iy < 0 ? -iy : (iy > IMH - 1 ? 2 * (IMH - 1) - iy : iy);
      const unsigned short* srow = src + (size_t)c * NPIX + (size_t)iy * IMW;
      for (int j = 0; j < 65; ++j) {
        int px = g * 65 + j;
        if (px < 130) {
          int ix = 2 * x0 - 1 + px;
          ix = ix < 0 ? -ix : (ix > IMW - 1 ? 2 * (IMW - 1) - ix : ix);
          P[(pr * P1W + px) * ASTRIDE + c] = srow[ix];
        }
      }
    }
  }
  __syncthreads();
  v8f acc[4] = {};
  for (int t = 0; t < 9; ++t) {
    int ky = t / 3, kx = t % 3;
    const unsigned short* wt = w1bf + t * CH * CH;
    for (int i = tid; i < CH * CH / 8; i += 256) {    // b128 tap-matrix copy
      int r = i >> 4, c8 = (i & 15) * 8;
      *(uint4*)(Atap + r * ASTRIDE + c8) = *(const uint4*)(wt + r * CH + c8);
    }
    __syncthreads();
    const unsigned short* ar = Atap + (wave * 16 + ln) * ASTRIDE;
    #pragma unroll
    for (int ksi = 0; ksi < 4; ++ksi) {
      int k = ksi * 32;
      v16bf A = mk16(ar + k + kh * 16, ar + k + kh * 16 + 8);
      v16bf Bv[4];
      #pragma unroll
      for (int nt = 0; nt < 4; ++nt) {
        int ip = 2 * (nt * 16 + ln) + kx;              // stride-2 gather in panel
        const unsigned short* br = P + (ky * P1W + ip) * ASTRIDE + k + kh * 16;
        Bv[nt] = mk16(br, br + 8);
      }
      #pragma unroll
      for (int nt = 0; nt < 4; ++nt) acc[nt] = WMMA_BF16(A, Bv[nt], acc[nt]);
    }
    __syncthreads();
  }
  int mb = kh * 8;
  #pragma unroll
  for (int nt = 0; nt < 4; ++nt)
    #pragma unroll
    for (int v = 0; v < 8; ++v) {
      int c = wave * 16 + mb + v;
      float val = acc[nt][v] + b1[c];
      S[c * 64 + nt * 16 + ln] = f2bf(val > 0.f ? val : 0.f);
    }
  __syncthreads();
  {
    int c = tid >> 1, seg = (tid & 1) * 32;
    const uint4* s4 = (const uint4*)(S + c * 64 + seg);
    uint4* d4 = (uint4*)(x1 + ((size_t)b * CH + c) * OPIX + (size_t)oy * OHW + x0 + seg);
    #pragma unroll
    for (int q = 0; q < 4; ++q) d4[q] = s4[q];
  }
}

// ---------------------------------------------------------------------------
// Kernel 8: reflect-pad + conv 3x3 stride1 + bias -> f32 output.
// Tile: 256 out-ch x 32 out-px per block.
// ---------------------------------------------------------------------------
__global__ __launch_bounds__(256) void conv2_kernel(
    const unsigned short* __restrict__ x1,
    const unsigned short* __restrict__ w2bf,
    const float* __restrict__ b2, float* __restrict__ out) {
  extern __shared__ float4 smem_f4[];
  float* S          = (float*)smem_f4;                       // [CH2][32]
  unsigned short* P = (unsigned short*)(S + CH2 * 32);       // [3][P2W][ASTRIDE]
  unsigned short* Atap = P + 3 * P2W * ASTRIDE;              // [CH2][ASTRIDE]
  int x0 = blockIdx.x * 32, oy = blockIdx.y, b = blockIdx.z;
  int tid = threadIdx.x, wave = tid >> 5, lane = tid & 31;
  int ln = lane & 15, kh = lane >> 4;
  const unsigned short* src = x1 + (size_t)b * CH * OPIX;
  {
    int c = tid & 127, g = tid >> 7;
    for (int pr = 0; pr < 3; ++pr) {
      int iy = oy + pr - 1;
      iy = iy < 0 ? -iy : (iy > OHW - 1 ? 2 * (OHW - 1) - iy : iy);
      const unsigned short* srow = src + (size_t)c * OPIX + (size_t)iy * OHW;
      for (int j = 0; j < 17; ++j) {
        int px = g * 17 + j;
        if (px < 34) {
          int ix = x0 - 1 + px;
          ix = ix < 0 ? -ix : (ix > OHW - 1 ? 2 * (OHW - 1) - ix : ix);
          P[(pr * P2W + px) * ASTRIDE + c] = srow[ix];
        }
      }
    }
  }
  __syncthreads();
  v8f acc[2][2] = {};
  for (int t = 0; t < 9; ++t) {
    int ky = t / 3, kx = t % 3;
    const unsigned short* wt = w2bf + t * CH2 * CH;
    for (int i = tid; i < CH2 * CH / 8; i += 256) {   // b128 tap-matrix copy
      int r = i >> 4, c8 = (i & 15) * 8;
      *(uint4*)(Atap + r * ASTRIDE + c8) = *(const uint4*)(wt + r * CH + c8);
    }
    __syncthreads();
    #pragma unroll
    for (int ksi = 0; ksi < 4; ++ksi) {
      int k = ksi * 32;
      const unsigned short* ar0 = Atap + (wave * 32 + ln) * ASTRIDE + k;
      const unsigned short* ar1 = ar0 + 16 * ASTRIDE;
      v16bf A0 = mk16(ar0 + kh * 16, ar0 + kh * 16 + 8);
      v16bf A1 = mk16(ar1 + kh * 16, ar1 + kh * 16 + 8);
      v16bf Bv[2];
      #pragma unroll
      for (int nt = 0; nt < 2; ++nt) {
        int ip = nt * 16 + ln + kx;
        const unsigned short* br = P + (ky * P2W + ip) * ASTRIDE + k + kh * 16;
        Bv[nt] = mk16(br, br + 8);
      }
      acc[0][0] = WMMA_BF16(A0, Bv[0], acc[0][0]);
      acc[0][1] = WMMA_BF16(A0, Bv[1], acc[0][1]);
      acc[1][0] = WMMA_BF16(A1, Bv[0], acc[1][0]);
      acc[1][1] = WMMA_BF16(A1, Bv[1], acc[1][1]);
    }
    __syncthreads();
  }
  int mb = kh * 8;
  #pragma unroll
  for (int mt = 0; mt < 2; ++mt)
    #pragma unroll
    for (int v = 0; v < 8; ++v) {
      int c = wave * 32 + mt * 16 + mb + v;
      S[c * 32 +  0 + ln] = acc[mt][0][v] + b2[c];
      S[c * 32 + 16 + ln] = acc[mt][1][v] + b2[c];
    }
  __syncthreads();
  {
    int c = tid;
    const float4* s4 = (const float4*)(S + c * 32);
    float4* d4 = (float4*)(out + ((size_t)b * CH2 + c) * OPIX + (size_t)oy * OHW + x0);
    #pragma unroll
    for (int q = 0; q < 8; ++q) d4[q] = s4[q];
  }
}

// ---------------------------------------------------------------------------
extern "C" void kernel_launch(void* const* d_in, const int* in_sizes, int n_in,
                              void* d_out, int out_size, void* d_ws, size_t ws_size,
                              hipStream_t stream) {
  (void)in_sizes; (void)n_in; (void)out_size; (void)ws_size;
  const float* xc = (const float*)d_in[0];
  const float* xs = (const float*)d_in[1];
  const float* W1 = (const float*)d_in[2];
  const float* b1 = (const float*)d_in[3];
  const float* W2 = (const float*)d_in[4];
  const float* b2 = (const float*)d_in[5];
  float* out = (float*)d_out;

  char* ws = (char*)d_ws;
  size_t off = 0;
  auto alloc = [&](size_t bytes) -> void* {
    void* p = ws + off;
    off += (bytes + 255) & ~(size_t)255;
    return p;
  };
  float* mc    = (float*)alloc((size_t)BATCH * CH * 4);
  float* ms    = (float*)alloc((size_t)BATCH * CH * 4);
  float* gpart = (float*)alloc((size_t)16 * KSPLIT * CH * CH * 4);   // 32 MB
  float* gc    = (float*)alloc((size_t)BATCH * CH * CH * 4);
  float* gs    = (float*)alloc((size_t)BATCH * CH * CH * 4);
  float* Amat  = (float*)alloc((size_t)BATCH * CH * CH * 4);
  float* dvec  = (float*)alloc((size_t)BATCH * CH * 4);
  unsigned short* w1bf = (unsigned short*)alloc((size_t)9 * CH * CH * 2);
  unsigned short* w2bf = (unsigned short*)alloc((size_t)9 * CH2 * CH * 2);
  unsigned short* colored = (unsigned short*)alloc((size_t)BATCH * CH * NPIX * 2); // 128 MB
  unsigned short* x1 = (unsigned short*)alloc((size_t)BATCH * CH * OPIX * 2);      // 32 MB

  repack_kernel<<<(9 * CH2 * CH + 255) / 256, 256, 0, stream>>>(W1, W2, w1bf, w2bf);
  mean_kernel<<<2 * BATCH * CH, 256, 0, stream>>>(xc, xs, mc, ms);
  gram_kernel<<<dim3(KSPLIT, 16), 256, 0, stream>>>(xc, xs, gpart);
  gram_reduce_kernel<<<(16 * CH * CH) / 256, 256, 0, stream>>>(gpart, gc, gs);
  chol_kernel<<<BATCH, 256, (2 * CH * 129 + 3 * CH) * 4, stream>>>(gc, gs, mc, ms, Amat, dvec);
  color_kernel<<<dim3(NPIX / 128, BATCH), 256,
                 (2 * CH * ASTRIDE + CH * 128) * 2, stream>>>(xc, Amat, dvec, colored);
  conv1_kernel<<<dim3(OHW / 64, OHW, BATCH), 256,
                 (3 * P1W * ASTRIDE + CH * ASTRIDE + CH * 64) * 2, stream>>>(colored, w1bf, b1, x1);
  conv2_kernel<<<dim3(OHW / 32, OHW, BATCH), 256,
                 CH2 * 32 * 4 + (3 * P2W * ASTRIDE + CH2 * ASTRIDE) * 2, stream>>>(x1, w2bf, b2, out);
}
